// TransformerMatchModule_73349451481579
// MI455X (gfx1250) — matrix operand order repaired and checked by reading, not compiled, not run on hardware
//
#include <hip/hip_runtime.h>

// ---------------------------------------------------------------------------
// TransformerMatchModule for MI455X (gfx1250, wave32, WMMA bf16 f32-acc).
// B=32, K=512, H=256, HEAD=4, W=126, d_self=64, d_lang=16.
// bf16 activations end-to-end; GEMM is a double-buffered pipeline with
// GLOBAL_LOAD_ASYNC_TO_LDS_B128 A-tile staging (s_wait_asynccnt) overlapped
// with v_wmma_f32_16x16x32_bf16; attention is flash-style with overlapped
// V-tile staging, fused dw bias + language mask.
// ---------------------------------------------------------------------------

#define BB 32
#define KK 512
#define HH 256
#define NHEAD 4
#define WW 126

typedef __attribute__((ext_vector_type(16))) __bf16 v16bf;
typedef __attribute__((ext_vector_type(8)))  __bf16 v8bf;
typedef __attribute__((ext_vector_type(8)))  float  v8f;
typedef __attribute__((ext_vector_type(4)))  int    v4i_t;

#if __has_builtin(__builtin_amdgcn_global_load_async_to_lds_b128) && \
    __has_builtin(__builtin_amdgcn_s_wait_asynccnt)
#define HAVE_ASYNC_LDS 1
typedef __attribute__((address_space(1))) v4i_t* g_v4i;  // global int4*
typedef __attribute__((address_space(3))) v4i_t* l_v4i;  // LDS int4*
#else
#define HAVE_ASYNC_LDS 0
#endif

static __device__ __forceinline__ v8f wmma_bf16(v16bf a, v16bf b, v8f c) {
  // (neg_a, A, neg_b, B, c_mod, C, reuse_a, reuse_b)
  return __builtin_amdgcn_wmma_f32_16x16x32_bf16(false, a, false, b, (short)0, c,
                                                 false, false);
}

// A-fragment (16-bit, 16x32): lane L holds row m=L&15; element j maps to
// k = (L>>4)*16 + j  -> two contiguous 8-half (16B) loads.
static __device__ __forceinline__ v16bf fragA2(v8bf x, v8bf y) {
  v16bf r;
#pragma unroll
  for (int i = 0; i < 8; ++i) { r[i] = x[i]; r[i + 8] = y[i]; }
  return r;
}

// B-fragment (16-bit, 32x16): lane L holds column n=L&15; element j maps to
// k = (L>>4)*16 + j  -> one contiguous 16-half (32B) load.
static __device__ __forceinline__ v16bf fragB(const __bf16* p) {
  return *(const v16bf*)p;
}

static __device__ __forceinline__ v8bf pack8(float4 a, float4 b) {
  v8bf r;
  r[0] = (__bf16)a.x; r[1] = (__bf16)a.y; r[2] = (__bf16)a.z; r[3] = (__bf16)a.w;
  r[4] = (__bf16)b.x; r[5] = (__bf16)b.y; r[6] = (__bf16)b.z; r[7] = (__bf16)b.w;
  return r;
}

// ---------------------------------------------------------------------------
// dw[b,i,j] = w_ij / sum_i' w_i'j,  w = 1/(dist(i,j)+0.01). Lane = column j.
// ---------------------------------------------------------------------------
__global__ __launch_bounds__(128) void dw_kernel(const float* __restrict__ centers,
                                                 float* __restrict__ dw) {
  const int b = blockIdx.y;
  const int j = blockIdx.x * 128 + threadIdx.x;
  const float* cb = centers + (size_t)b * KK * 3;
  const float xj = cb[j * 3 + 0], yj = cb[j * 3 + 1], zj = cb[j * 3 + 2];
  float sum = 0.f;
  for (int i = 0; i < KK; ++i) {
    float dx = cb[i * 3 + 0] - xj;
    float dy = cb[i * 3 + 1] - yj;
    float dz = cb[i * 3 + 2] - zj;
    float d = sqrtf(dx * dx + dy * dy + dz * dz);
    sum += 1.0f / (d + 0.01f);
  }
  const float inv = 1.0f / sum;
  float* out = dw + (size_t)b * KK * KK;
  for (int i = 0; i < KK; ++i) {
    float dx = cb[i * 3 + 0] - xj;
    float dy = cb[i * 3 + 1] - yj;
    float dz = cb[i * 3 + 2] - zj;
    float d = sqrtf(dx * dx + dy * dy + dz * dz);
    out[(size_t)i * KK + j] = (1.0f / (d + 0.01f)) * inv;
  }
}

// ---------------------------------------------------------------------------
// GEMM: C[M,N] = epi(A[M,Kd] @ W + bias). Compile-time specialized:
//   ABF: A is bf16 (async LDS staging) vs f32 (convert while staging).
//   WT:  W[N][Kd] conv-style (1) vs W[Kd][N] x@w-style (0).
// Runtime: epi 0=bias, 1=+relu, 2=+BN+PReLU (pstride 0 => scalar alpha);
//          out_bf16 selects store type.
// 256 threads / 8 waves; block tile 64x64; k-step 32; double-buffered LDS
// pipeline: one barrier per k-step, next tile staged during WMMA.
// ---------------------------------------------------------------------------
template <int ABF, int WT>
__global__ __launch_bounds__(256) void gemm_kern(
    const void* __restrict__ Av, const float* __restrict__ Wm,
    const float* __restrict__ bias, const float* __restrict__ bn_g,
    const float* __restrict__ bn_b, const float* __restrict__ bn_m,
    const float* __restrict__ bn_v, const float* __restrict__ prelu, int pstride,
    void* __restrict__ Cv, int M, int N, int Kd, int epi, int out_bf16) {
  __shared__ __align__(32) __bf16 Alds[2][64][32];
  __shared__ __align__(32) __bf16 Blds[2][64][32];  // [n][k]
  const int tid = threadIdx.x;
  const int lane = tid & 31;
  const int wave = tid >> 5;
  const int wm = wave & 3, wn = wave >> 2;
  const int lo16 = lane & 15, hi = lane >> 4;
  const int m0 = blockIdx.x * 64;
  const int n0 = blockIdx.y * 64;
  v8f acc[2] = {};
  const v8bf zero8 = {};

  auto stage = [&](int k0, int buf) {
    {  // A tile 64x32 bf16: row=t/4, kgroup=(t%4)*8 (16B per thread).
      const int row = tid >> 2, kg = (tid & 3) * 8;
      const bool ok = (m0 + row) < M;
      if constexpr (ABF) {
        const __bf16* src = (const __bf16*)Av + (size_t)(m0 + row) * Kd + k0 + kg;
        if (ok) {
#if HAVE_ASYNC_LDS
          v4i_t* sg = const_cast<v4i_t*>(reinterpret_cast<const v4i_t*>(src));
          v4i_t* dl = reinterpret_cast<v4i_t*>(&Alds[buf][row][kg]);
          __builtin_amdgcn_global_load_async_to_lds_b128((g_v4i)sg, (l_v4i)dl,
                                                         0, 0);
#else
          *(v8bf*)&Alds[buf][row][kg] = *(const v8bf*)src;
#endif
        } else {
          *(v8bf*)&Alds[buf][row][kg] = zero8;
        }
      } else {
        const float* src = (const float*)Av + (size_t)(m0 + row) * Kd + k0 + kg;
        v8bf pk = zero8;
        if (ok) {
          float4 f0 = *(const float4*)src;
          float4 f1 = *(const float4*)(src + 4);
          pk = pack8(f0, f1);
        }
        *(v8bf*)&Alds[buf][row][kg] = pk;
      }
    }
    if constexpr (WT) {  // W[N][Kd], k contiguous -> straight b128 store
      const int n = tid >> 2, kg = (tid & 3) * 8;
      const float* src = Wm + (size_t)(n0 + n) * Kd + k0 + kg;
      v8bf pk = zero8;
      if ((n0 + n) < N) {
        float4 f0 = *(const float4*)src;
        float4 f1 = *(const float4*)(src + 4);
        pk = pack8(f0, f1);
      }
      *(v8bf*)&Blds[buf][n][kg] = pk;
    } else {  // W[Kd][N], n contiguous -> transpose into LDS
      const int k = tid >> 3, ng = (tid & 7) * 8;
      if ((n0 + ng) < N) {  // N % 64 == 0 -> whole 8-group in or out
        const float* src = Wm + (size_t)(k0 + k) * N + n0 + ng;
        float4 f0 = *(const float4*)src;
        float4 f1 = *(const float4*)(src + 4);
        v8bf pk = pack8(f0, f1);
#pragma unroll
        for (int i = 0; i < 8; ++i) Blds[buf][ng + i][k] = pk[i];
      } else {
#pragma unroll
        for (int i = 0; i < 8; ++i) Blds[buf][ng + i][k] = (__bf16)0.0f;
      }
    }
  };

  stage(0, 0);  // prologue
  const int niter = Kd >> 5;
  for (int i = 0; i < niter; ++i) {
    const int buf = i & 1;
#if HAVE_ASYNC_LDS
    if constexpr (ABF) __builtin_amdgcn_s_wait_asynccnt(0);
#endif
    __syncthreads();  // tile i visible to all waves; tile i-1 consumers done
    if (i + 1 < niter) stage((i + 1) << 5, buf ^ 1);  // overlap with WMMA

    const int am = wm * 16 + lo16;
    v16bf af = fragA2(*(const v8bf*)&Alds[buf][am][hi * 16],
                      *(const v8bf*)&Alds[buf][am][hi * 16 + 8]);
    const int nb = wn * 32 + lo16;
    v16bf b0 = fragB(&Blds[buf][nb][hi * 16]);
    v16bf b1 = fragB(&Blds[buf][nb + 16][hi * 16]);
    acc[0] = wmma_bf16(af, b0, acc[0]);
    acc[1] = wmma_bf16(af, b1, acc[1]);
  }

#pragma unroll
  for (int t = 0; t < 2; ++t) {
    const int n = n0 + wn * 32 + t * 16 + lo16;
    if (n >= N) continue;
    const float bv = bias ? bias[n] : 0.f;
    float sc = 1.f, sh = 0.f, al = 1.f;
    if (epi == 2) {
      sc = rsqrtf(bn_v[n] + 1e-5f) * bn_g[n];
      sh = bn_b[n] - bn_m[n] * sc;
      al = prelu[n * pstride];
    }
#pragma unroll
    for (int r = 0; r < 8; ++r) {
      const int m = m0 + wm * 16 + hi * 8 + r;  // C layout: row = r + 8*(L>>4)
      if (m >= M) continue;
      float x = acc[t][r] + bv;
      if (epi == 1) x = fmaxf(x, 0.f);
      else if (epi == 2) { x = x * sc + sh; x = (x > 0.f) ? x : al * x; }
      if (out_bf16) ((__bf16*)Cv)[(size_t)m * N + n] = (__bf16)x;
      else ((float*)Cv)[(size_t)m * N + n] = x;
    }
  }
}

// ---------------------------------------------------------------------------
// Flash attention over bf16 Q/K/V (B, N, NHEAD*D); bf16 output.
// grid=(ceil(Nq/64), B*NHEAD), block=128 (wave owns 16 q-rows). Key chunks
// of 32, double-buffered V staging overlapped with score WMMA + softmax:
// direct-global 32B K B-fragments; +scale +dw, mask; online softmax;
// P->LDS->A-frag; WMMA P@V. OOB keys handled by masking (slabs have 2x
// headroom; zeroed Q upper-k kills neighbor-head K for D=16).
// ---------------------------------------------------------------------------
__global__ __launch_bounds__(128) void attn_kernel(
    const __bf16* __restrict__ Qb, const __bf16* __restrict__ Kb,
    const __bf16* __restrict__ Vb, __bf16* __restrict__ Ob,
    const float* __restrict__ addw, const unsigned char* __restrict__ mask,
    int Nq, int Nk, int D, int nheads, float scale) {
  __shared__ __align__(32) __bf16 Vtlds[2][64][32];  // [d][key]
  __shared__ __align__(32) __bf16 Plds[4][16][32];
  const int tid = threadIdx.x, lane = tid & 31, wave = tid >> 5;
  const int lo16 = lane & 15, hi = lane >> 4;
  const int bh = blockIdx.y;
  const int b = bh / nheads, head = bh % nheads;
  const int q0 = blockIdx.x * 64 + wave * 16;
  const int ldq = nheads * D;
  const int DP = (D + 31) & ~31;   // 32 or 64
  const int nDC = DP >> 5;         // score d-chunks of 32
  const int nDT = DP >> 4;         // output d-tiles of 16
  const __bf16* Qbase = Qb + (size_t)b * Nq * ldq + head * D;
  const __bf16* Kbase = Kb + (size_t)b * Nk * ldq + head * D;
  const __bf16* Vbase = Vb + (size_t)b * Nk * ldq + head * D;

  // Q fragments (vector loads; zero chunks with d >= D).
  const int qrow = q0 + lo16;
  const __bf16* qp = Qbase + (size_t)qrow * ldq;
  const v8bf z8 = {};
  v16bf qf[2];
#pragma unroll
  for (int c = 0; c < 2; ++c) {
    v8bf lo = ((c * 32 + hi * 16 < D) && qrow < Nq) ? *(const v8bf*)(qp + c * 32 + hi * 16) : z8;
    v8bf hi8 = ((c * 32 + hi * 16 + 8 < D) && qrow < Nq)
                   ? *(const v8bf*)(qp + c * 32 + hi * 16 + 8) : z8;
    qf[c] = fragA2(lo, hi8);
  }

  // Stage V chunk transposed: thread -> (key=t/4, dgroup=(t%4)*16), 32B load.
  auto stageV = [&](int kb, int buf) {
    const int key = tid >> 2, dg = (tid & 3) * 16;
    const __bf16* vp = Vbase + (size_t)(kb + key) * ldq + dg;
    v8bf a = *(const v8bf*)vp;
    v8bf c2 = *(const v8bf*)(vp + 8);
#pragma unroll
    for (int i = 0; i < 8; ++i) {
      Vtlds[buf][dg + i][key] = a[i];
      Vtlds[buf][dg + 8 + i][key] = c2[i];
    }
  };

  v8f oacc[4] = {};
  float rm[8], rl[8];
#pragma unroll
  for (int r = 0; r < 8; ++r) { rm[r] = -1e30f; rl[r] = 0.f; }

  const int nchunks = ((Nk + 31) & ~31) >> 5;
  stageV(0, 0);  // prologue
  for (int c = 0; c < nchunks; ++c) {
    const int kb = c << 5;
    const int buf = c & 1;
    __syncthreads();  // V chunk c visible; previous PV readers of buf^1 done
    if (c + 1 < nchunks) stageV(kb + 32, buf ^ 1);  // overlap with scores

    // Scores (16q x 32k): K fragments straight from global (32B each).
    v8f s0 = {}, s1 = {};
    const __bf16* kp0 = Kbase + (size_t)(kb + lo16) * ldq;
    const __bf16* kp1 = Kbase + (size_t)(kb + 16 + lo16) * ldq;
#pragma unroll
    for (int cc = 0; cc < 2; ++cc) {
      if (cc < nDC) {
        v16bf k0f = fragB(kp0 + cc * 32 + hi * 16);
        v16bf k1f = fragB(kp1 + cc * 32 + hi * 16);
        s0 = wmma_bf16(qf[cc], k0f, s0);
        s1 = wmma_bf16(qf[cc], k1f, s1);
      }
    }

    // scale, +dw bias (self-attn only; all rows/keys in range), mask
    float p0v[8], p1v[8];
    const int k0g = kb + lo16, k1g = kb + 16 + lo16;
    const bool ok0 = (k0g < Nk) && (!mask || mask[(size_t)b * Nk + k0g] != 0);
    const bool ok1 = (k1g < Nk) && (!mask || mask[(size_t)b * Nk + k1g] != 0);
#pragma unroll
    for (int r = 0; r < 8; ++r) {
      float a0 = s0[r] * scale, a1 = s1[r] * scale;
      if (addw) {
        const float* awr = addw + ((size_t)b * Nq + (q0 + hi * 8 + r)) * Nk;
        a0 += awr[k0g];
        a1 += awr[k1g];
      }
      p0v[r] = ok0 ? a0 : -1e30f;
      p1v[r] = ok1 ? a1 : -1e30f;
    }

    // Online softmax; P -> per-wave LDS tile (C->A relayout).
#pragma unroll
    for (int r = 0; r < 8; ++r) {
      float tmax = fmaxf(p0v[r], p1v[r]);
#pragma unroll
      for (int mk = 8; mk >= 1; mk >>= 1)
        tmax = fmaxf(tmax, __shfl_xor(tmax, mk, 32));  // 16-lane half reduce
      const float mnew = fmaxf(rm[r], tmax);
      const float corr = __expf(rm[r] - mnew);
      const float e0 = __expf(p0v[r] - mnew);
      const float e1 = __expf(p1v[r] - mnew);
      float rs = e0 + e1;
#pragma unroll
      for (int mk = 8; mk >= 1; mk >>= 1) rs += __shfl_xor(rs, mk, 32);
      rl[r] = rl[r] * corr + rs;
      rm[r] = mnew;
#pragma unroll
      for (int t = 0; t < 4; ++t) oacc[t][r] = oacc[t][r] * corr;
      Plds[wave][hi * 8 + r][lo16] = (__bf16)e0;
      Plds[wave][hi * 8 + r][16 + lo16] = (__bf16)e1;
    }
    __syncthreads();  // uniform trip count; publishes cross-lane P

    // O += P @ V
    v16bf pf = fragA2(*(const v8bf*)&Plds[wave][lo16][hi * 16],
                      *(const v8bf*)&Plds[wave][lo16][hi * 16 + 8]);
#pragma unroll
    for (int t = 0; t < 4; ++t) {
      if (t < nDT) {
        const int dcol = t * 16 + lo16;
        v16bf vf = fragB(&Vtlds[buf][dcol][hi * 16]);
        oacc[t] = wmma_bf16(pf, vf, oacc[t]);
      }
    }
  }

  // Normalize and store bf16 (feeds the O-projection GEMM's async-A path).
#pragma unroll
  for (int r = 0; r < 8; ++r) {
    const int qr = q0 + hi * 8 + r;
    if (qr >= Nq) continue;
    const float inv = (rl[r] > 0.f) ? (1.0f / rl[r]) : 0.f;
#pragma unroll
    for (int t = 0; t < 4; ++t) {
      const int d = t * 16 + lo16;
      if (d < D)
        Ob[((size_t)b * Nq + qr) * ldq + head * D + d] = (__bf16)(oacc[t][r] * inv);
    }
  }
}

// ---------------------------------------------------------------------------
// LayerNorm over 256: x (f32) [+ res (bf16)] -> out (bf16). Wave per row;
// lane owns 8 contiguous columns (float4/v8bf vector accesses).
// ---------------------------------------------------------------------------
__global__ __launch_bounds__(128) void ln_kernel(
    const float* __restrict__ x, const __bf16* __restrict__ res,
    const float* __restrict__ g, const float* __restrict__ beta,
    __bf16* __restrict__ out, int rows) {
  const int row = blockIdx.x * 4 + (threadIdx.x >> 5);
  const int lane = threadIdx.x & 31;
  if (row >= rows) return;
  const float* xr = x + (size_t)row * HH + lane * 8;
  float v[8];
  float4 f0 = *(const float4*)xr;
  float4 f1 = *(const float4*)(xr + 4);
  v[0] = f0.x; v[1] = f0.y; v[2] = f0.z; v[3] = f0.w;
  v[4] = f1.x; v[5] = f1.y; v[6] = f1.z; v[7] = f1.w;
  if (res) {
    v8bf rv = *(const v8bf*)(res + (size_t)row * HH + lane * 8);
#pragma unroll
    for (int t = 0; t < 8; ++t) v[t] += (float)rv[t];
  }
  float s = 0.f;
#pragma unroll
  for (int t = 0; t < 8; ++t) s += v[t];
#pragma unroll
  for (int mk = 16; mk >= 1; mk >>= 1) s += __shfl_xor(s, mk, 32);
  const float mean = s * (1.0f / 256.0f);
  float vs = 0.f;
#pragma unroll
  for (int t = 0; t < 8; ++t) { const float d = v[t] - mean; vs += d * d; }
#pragma unroll
  for (int mk = 16; mk >= 1; mk >>= 1) vs += __shfl_xor(vs, mk, 32);
  const float inv = rsqrtf(vs * (1.0f / 256.0f) + 1e-5f);
  const float* gp = g + lane * 8;
  const float* bp = beta + lane * 8;
  float4 g0 = *(const float4*)gp, g1 = *(const float4*)(gp + 4);
  float4 b0 = *(const float4*)bp, b1 = *(const float4*)(bp + 4);
  const float gv[8] = {g0.x, g0.y, g0.z, g0.w, g1.x, g1.y, g1.z, g1.w};
  const float bv[8] = {b0.x, b0.y, b0.z, b0.w, b1.x, b1.y, b1.z, b1.w};
  v8bf o;
#pragma unroll
  for (int t = 0; t < 8; ++t) o[t] = (__bf16)((v[t] - mean) * inv * gv[t] + bv[t]);
  *(v8bf*)(out + (size_t)row * HH + lane * 8) = o;
}

// Final 1xH head: out[row] = dot(X[row] (bf16), w3) + b3. Wave per row.
__global__ __launch_bounds__(128) void rowdot_kernel(
    const __bf16* __restrict__ X, const float* __restrict__ w,
    const float* __restrict__ b, float* __restrict__ out, int rows) {
  const int row = blockIdx.x * 4 + (threadIdx.x >> 5);
  const int lane = threadIdx.x & 31;
  if (row >= rows) return;
  v8bf xv = *(const v8bf*)(X + (size_t)row * HH + lane * 8);
  const float* wp = w + lane * 8;
  float4 w0 = *(const float4*)wp, w1 = *(const float4*)(wp + 4);
  const float wv[8] = {w0.x, w0.y, w0.z, w0.w, w1.x, w1.y, w1.z, w1.w};
  float s = 0.f;
#pragma unroll
  for (int t = 0; t < 8; ++t) s += (float)xv[t] * wv[t];
#pragma unroll
  for (int mk = 16; mk >= 1; mk >>= 1) s += __shfl_xor(s, mk, 32);
  if (lane == 0) out[row] = s + b[0];
}

// ---------------------------------------------------------------------------
extern "C" void kernel_launch(void* const* d_in, const int* in_sizes, int n_in,
                              void* d_out, int out_size, void* d_ws, size_t ws_size,
                              hipStream_t stream) {
  (void)in_sizes; (void)n_in; (void)out_size; (void)ws_size;
  auto F = [&](int i) { return (const float*)d_in[i]; };
  const float* geo_centers = F(0);
  const float* geo_feats = F(1);
  const float* lang_hidden = F(3);
  const unsigned char* lang_masks = (const unsigned char*)d_in[4];  // bool(B,W)
  const float *f_w1 = F(6), *f_b1 = F(7), *f_g = F(8), *f_bb = F(9),
              *f_m = F(10), *f_v = F(11), *f_pr = F(12), *f_w2 = F(13), *f_b2 = F(14);
  const float *l_w = F(15), *l_b = F(16), *l_g = F(17), *l_bb = F(18);
  const float *al_wq = F(19), *al_bq = F(20), *al_wk = F(21), *al_bk = F(22),
              *al_wv = F(23), *al_bv = F(24), *al_wo = F(25), *al_bo = F(26),
              *al_g = F(27), *al_bb = F(28);
  const float *as_wq = F(29), *as_bq = F(30), *as_wk = F(31), *as_bk = F(32),
              *as_wv = F(33), *as_bv = F(34), *as_wo = F(35), *as_bo = F(36),
              *as_g = F(37), *as_bb = F(38);
  const float *ac_wq = F(39), *ac_bq = F(40), *ac_wk = F(41), *ac_bk = F(42),
              *ac_wv = F(43), *ac_bv = F(44), *ac_wo = F(45), *ac_bo = F(46),
              *ac_g = F(47), *ac_bb = F(48);
  const float *m_w1 = F(49), *m_b1 = F(50), *m1g = F(51), *m1b = F(52),
              *m1m = F(53), *m1v = F(54), *m_p1 = F(55), *m_w2 = F(56),
              *m_b2 = F(57), *m2g = F(58), *m2b = F(59), *m2m = F(60),
              *m2v = F(61), *m_p2 = F(62), *m_w3 = F(63), *m_b3 = F(64);

  // Workspace slabs (counted in floats; bf16 views use half a slab -> 2x
  // headroom for guard-free OOB key reads). ~168 MB total.
  float* ws = (float*)d_ws;
  size_t off = 0;
  auto alloc = [&](size_t n) { float* p = ws + off; off += n; return p; };
  const size_t NB = (size_t)BB * KK * HH;   // 4,194,304
  const size_t NL = (size_t)BB * WW * HH;   // 1,032,192
  const size_t NS = (size_t)BB * WW * 64;   //   258,048
  float* dw = alloc((size_t)BB * KK * KK);
  __bf16* S0h = (__bf16*)alloc(NB);    // state ping
  __bf16* S1h = (__bf16*)alloc(NB);    // state pong
  __bf16* TQ = (__bf16*)alloc(NB);     // q-proj / match intermediates
  __bf16* TK = (__bf16*)alloc(NB);     // k-proj / match intermediates
  __bf16* TV = (__bf16*)alloc(NB);     // v-proj
  __bf16* TA = (__bf16*)alloc(NB);     // attention output (bf16)
  float* TO = alloc(NB);               // o-proj output (f32, LN input)
  __bf16* Llf = (__bf16*)alloc(NL);    // lang LN state; later cross-K buffer
  __bf16* Ltmp = (__bf16*)alloc(NL);   // LF state (post lang-MHA)
  float* Lo = alloc(NL);               // lang FC / o-proj f32 tmp; cross-V buf
  __bf16* Lq = (__bf16*)alloc(NS);
  __bf16* Lk = (__bf16*)alloc(NS);
  __bf16* Lv = (__bf16*)alloc(NS);
  __bf16* La = (__bf16*)alloc(NS);

  auto gemm = [&](const void* A, int abf, const float* Wm, const float* bias,
                  void* C, int M, int N, int Kd, int wT, int epi, int obf,
                  const float* g, const float* bta, const float* mn,
                  const float* vr, const float* pr, int ps) {
    dim3 grid((M + 63) / 64, (N + 63) / 64);
    if (abf) {
      if (wT)
        gemm_kern<1, 1><<<grid, 256, 0, stream>>>(A, Wm, bias, g, bta, mn, vr,
                                                  pr, ps, C, M, N, Kd, epi, obf);
      else
        gemm_kern<1, 0><<<grid, 256, 0, stream>>>(A, Wm, bias, g, bta, mn, vr,
                                                  pr, ps, C, M, N, Kd, epi, obf);
    } else {
      if (wT)
        gemm_kern<0, 1><<<grid, 256, 0, stream>>>(A, Wm, bias, g, bta, mn, vr,
                                                  pr, ps, C, M, N, Kd, epi, obf);
      else
        gemm_kern<0, 0><<<grid, 256, 0, stream>>>(A, Wm, bias, g, bta, mn, vr,
                                                  pr, ps, C, M, N, Kd, epi, obf);
    }
  };
  auto lnorm = [&](const float* x, const __bf16* res, const float* g,
                   const float* bt, __bf16* out, int rows) {
    ln_kernel<<<(rows + 3) / 4, 128, 0, stream>>>(x, res, g, bt, out, rows);
  };
  auto mha = [&](const __bf16* qin, const __bf16* kvin, int Nq, int Nk, int D,
                 const float* wq, const float* bq, const float* wk,
                 const float* bk, const float* wv, const float* bv,
                 const float* wo, const float* bo, const float* lg,
                 const float* lb, const float* addw, const unsigned char* msk,
                 float scl, __bf16* qb, __bf16* kb, __bf16* vb, __bf16* ab,
                 float* ob, __bf16* out) {
    const int Mq = BB * Nq, Mk = BB * Nk, HD = NHEAD * D;
    gemm(qin, 1, wq, bq, qb, Mq, HD, HH, 0, 0, 1, nullptr, nullptr, nullptr, nullptr, nullptr, 0);
    gemm(kvin, 1, wk, bk, kb, Mk, HD, HH, 0, 0, 1, nullptr, nullptr, nullptr, nullptr, nullptr, 0);
    gemm(kvin, 1, wv, bv, vb, Mk, HD, HH, 0, 0, 1, nullptr, nullptr, nullptr, nullptr, nullptr, 0);
    dim3 ag((Nq + 63) / 64, BB * NHEAD);
    attn_kernel<<<ag, 128, 0, stream>>>(qb, kb, vb, ab, addw, msk, Nq, Nk, D,
                                        NHEAD, scl);
    gemm(ab, 1, wo, bo, ob, Mq, HH, HD, 0, 0, 0, nullptr, nullptr, nullptr, nullptr, nullptr, 0);
    lnorm(ob, qin, lg, lb, out, Mq);
  };

  // 1) distance prior
  dw_kernel<<<dim3(KK / 128, BB), 128, 0, stream>>>(geo_centers, dw);

  // 2) feature MLP (f32 entry): conv+BN+PReLU -> bf16 x1; conv -> S0h state
  gemm(geo_feats, 0, f_w1, f_b1, TQ, BB * KK, HH, HH, 1, 2, 1, f_g, f_bb, f_m, f_v, f_pr, 1);
  gemm(TQ, 1, f_w2, f_b2, S0h, BB * KK, HH, HH, 1, 0, 1, nullptr, nullptr, nullptr, nullptr, nullptr, 0);

  // 3) language (f32 entry): FC+ReLU -> f32 Lo; LN -> Llf; masked MHA -> Ltmp
  gemm(lang_hidden, 0, l_w, l_b, Lo, BB * WW, HH, HH, 0, 1, 0, nullptr, nullptr, nullptr, nullptr, nullptr, 0);
  lnorm(Lo, nullptr, l_g, l_bb, Llf, BB * WW);
  mha(Llf, Llf, WW, WW, 16, al_wq, al_bq, al_wk, al_bk, al_wv, al_bv, al_wo,
      al_bo, al_g, al_bb, nullptr, lang_masks, 0.25f, Lq, Lk, Lv, La, Lo, Ltmp);
  const __bf16* LF = Ltmp;

  // 4) self0 (+dw) -> S1h ; cross0 (masked, kv=LF) -> S0h
  mha(S0h, S0h, KK, KK, 64, as_wq, as_bq, as_wk, as_bk, as_wv, as_bv, as_wo,
      as_bo, as_g, as_bb, dw, nullptr, 0.125f, TQ, TK, TV, TA, TO, S1h);
  mha(S1h, LF, KK, WW, 64, ac_wq, ac_bq, ac_wk, ac_bk, ac_wv, ac_bv, ac_wo,
      ac_bo, ac_g, ac_bb, nullptr, lang_masks, 0.125f, TQ, Llf, (__bf16*)Lo, TA,
      TO, S0h);

  // 5) self1 -> S1h ; cross1 -> S0h (layer-1 weight offsets)
  const size_t WO = (size_t)HH * HH, BO = HH;
  mha(S0h, S0h, KK, KK, 64, as_wq + WO, as_bq + BO, as_wk + WO, as_bk + BO,
      as_wv + WO, as_bv + BO, as_wo + WO, as_bo + BO, as_g + BO, as_bb + BO,
      dw, nullptr, 0.125f, TQ, TK, TV, TA, TO, S1h);
  mha(S1h, LF, KK, WW, 64, ac_wq + WO, ac_bq + BO, ac_wk + WO, ac_bk + BO,
      ac_wv + WO, ac_bv + BO, ac_wo + WO, ac_bo + BO, ac_g + BO, ac_bb + BO,
      nullptr, lang_masks, 0.125f, TQ, Llf, (__bf16*)Lo, TA, TO, S0h);

  // 6) match head: two conv+BN+PReLU(scalar) GEMMs (bf16), 1xH dot -> d_out
  gemm(S0h, 1, m_w1, m_b1, TQ, BB * KK, HH, HH, 1, 2, 1, m1g, m1b, m1m, m1v, m_p1, 0);
  gemm(TQ, 1, m_w2, m_b2, TK, BB * KK, HH, HH, 1, 2, 1, m2g, m2b, m2m, m2v, m_p2, 0);
  rowdot_kernel<<<(BB * KK + 3) / 4, 128, 0, stream>>>(TK, m_w3, m_b3,
                                                       (float*)d_out, BB * KK);
}
